// GraphUNet_55808805045040
// MI455X (gfx1250) — matrix operand order repaired and checked by reading, not compiled, not run on hardware
//
#include <hip/hip_runtime.h>
#include <hip/hip_bf16.h>
#include <math.h>

typedef __attribute__((ext_vector_type(16))) _Float16 v16h;
typedef __attribute__((ext_vector_type(8)))  _Float16 v8h;
typedef __attribute__((ext_vector_type(8)))  float    v8f;

static constexpr int NN    = 300000;
static constexpr int NE    = 600000;
static constexpr int HIDC  = 128;

// gfx1250 async global->LDS copy (ASYNCcnt-tracked), with safe fallback
#if defined(__HIP_DEVICE_COMPILE__) && __has_builtin(__builtin_amdgcn_global_load_async_to_lds_b128)
#define HAVE_ASYNC_LDS 1
typedef int vi4 __attribute__((vector_size(16)));
typedef __attribute__((address_space(1))) vi4 gvi4;   // global AS vector-int
typedef __attribute__((address_space(3))) vi4 lvi4;   // LDS    AS vector-int
#else
#define HAVE_ASYNC_LDS 0
#endif

// ---------------------------------------------------------------- utilities
__device__ __forceinline__ float gelu_f(float x) {
    return 0.5f * x * (1.0f + erff(x * 0.70710678118654752440f));
}
// order-preserving float -> uint (monotonic increasing)
__device__ __forceinline__ unsigned fenc(float f) {
    unsigned u = __float_as_uint(f);
    return (u & 0x80000000u) ? ~u : (u | 0x80000000u);
}
__device__ __forceinline__ float fdec(unsigned u) {
    unsigned v = (u & 0x80000000u) ? (u & 0x7FFFFFFFu) : ~u;
    return __uint_as_float(v);
}

__global__ void fill_u32_kernel(unsigned* __restrict__ p, unsigned v, long long n) {
    long long i = (long long)blockIdx.x * 256 + threadIdx.x;
    if (i < n) p[i] = v;
}

// ---------------------------------------------------------------- embedding
// x0[i] = concat( node_emb[type[i]] (128), LN(pos[i]) @ proj_W + proj_b (32) )
__global__ void embed_kernel(const int* __restrict__ types, const float* __restrict__ posi,
                             const float* __restrict__ emb, const float* __restrict__ lng,
                             const float* __restrict__ lnb, const float* __restrict__ pW,
                             const float* __restrict__ pb, float* __restrict__ x0, int n) {
    int i = blockIdx.x * 256 + threadIdx.x;
    if (i >= n) return;
    int t = types[i];
    const float* er = emb + (size_t)t * 128;
    float* xr = x0 + (size_t)i * 160;
    for (int c = 0; c < 128; c += 4) *(float4*)(xr + c) = *(const float4*)(er + c);
    float p0 = posi[2 * i], p1 = posi[2 * i + 1];
    float mu = 0.5f * (p0 + p1);
    float var = 0.5f * ((p0 - mu) * (p0 - mu) + (p1 - mu) * (p1 - mu));
    float inv = rsqrtf(var + 1e-5f);
    float q0 = (p0 - mu) * inv * lng[0] + lnb[0];
    float q1 = (p1 - mu) * inv * lng[1] + lnb[1];
    for (int j = 0; j < 32; ++j) xr[128 + j] = q0 * pW[j] + q1 * pW[32 + j] + pb[j];
}

// ---------------------------------------------------------------- WMMA GEMM
// H[n,128] = X[n,K] @ W[K,128]   (fp32 in/out, f16 WMMA, fp32 accumulate)
// Block: 256 threads = 8 waves; each wave owns one 16-col N tile; block loops
// 16 M-tiles (256 rows). W staged transposed (f16) in LDS once per block;
// X tile staged fp32 in LDS via async global->LDS B128 copies (no VGPR
// round-trip), converted to f16 while building A fragments.
__global__ __launch_bounds__(256) void gemm_wmma_kernel(
        const float* __restrict__ X, int ldx, int K,
        const float* __restrict__ W, float* __restrict__ H, int n) {
    __shared__ alignas(32) _Float16 sW[128 * 160];   // [col][k], f16
    __shared__ alignas(32) float    sX[16 * 160];    // [m][k],   f32
    const int tid  = threadIdx.x;
    const int lane = tid & 31;
    const int wv   = tid >> 5;        // 0..7 -> N tile
    const int hh   = lane >> 4;       // lane half
    const int ln   = lane & 15;
    const int col0 = wv * 16;

    for (int idx = tid; idx < K * 128; idx += 256) {
        int k = idx >> 7, c = idx & 127;
        sW[c * K + k] = (_Float16)W[idx];
    }
    __syncthreads();

    const int kchunks = K >> 2;               // float4 chunks per row
    for (int mt = 0; mt < 16; ++mt) {
        int rowbase = blockIdx.x * 256 + mt * 16;
        if (rowbase >= n) break;
        // stage 16 x K fp32 tile: 16B per lane per chunk
        for (int idx = tid; idx < 16 * kchunks; idx += 256) {
            int m  = idx / kchunks;
            int c4 = idx - m * kchunks;
            int row = rowbase + m;
            if (row >= n) row = n - 1;        // clamp; rows >= n never stored
            const float* gl = X + (size_t)row * ldx + c4 * 4;
            float* ld = &sX[m * K + c4 * 4];
            int prow = row + 16; if (prow >= n) prow = n - 1;
            __builtin_prefetch(X + (size_t)prow * ldx + c4 * 4, 0, 0);
#if HAVE_ASYNC_LDS
            __builtin_amdgcn_global_load_async_to_lds_b128(
                    (gvi4*)(void*)gl, (lvi4*)(void*)ld, 0, 0);
#else
            *(float4*)ld = *(const float4*)gl;
#endif
        }
#if HAVE_ASYNC_LDS
#if __has_builtin(__builtin_amdgcn_s_wait_asynccnt)
        __builtin_amdgcn_s_wait_asynccnt(0);
#else
        asm volatile("s_wait_asynccnt 0" ::: "memory");
#endif
#endif
        __syncthreads();

        v8f acc = {};
        for (int k0 = 0; k0 < K; k0 += 32) {
            // A fragment: lane half hh covers K=[hh*16, hh*16+16)
            const float* ar = &sX[ln * K + k0 + hh * 16];
            float4 a0 = *(const float4*)(ar);
            float4 a1 = *(const float4*)(ar + 4);
            float4 a2 = *(const float4*)(ar + 8);
            float4 a3 = *(const float4*)(ar + 12);
            v16h a;
            a[0]  = (_Float16)a0.x; a[1]  = (_Float16)a0.y;
            a[2]  = (_Float16)a0.z; a[3]  = (_Float16)a0.w;
            a[4]  = (_Float16)a1.x; a[5]  = (_Float16)a1.y;
            a[6]  = (_Float16)a1.z; a[7]  = (_Float16)a1.w;
            a[8]  = (_Float16)a2.x; a[9]  = (_Float16)a2.y;
            a[10] = (_Float16)a2.z; a[11] = (_Float16)a2.w;
            a[12] = (_Float16)a3.x; a[13] = (_Float16)a3.y;
            a[14] = (_Float16)a3.z; a[15] = (_Float16)a3.w;
            // B fragment: lane half hh covers K=[hh*16, hh*16+16), col col0+ln
            v16h b = *(const v16h*)&sW[(size_t)(col0 + ln) * K + k0 + hh * 16];
            acc = __builtin_amdgcn_wmma_f32_16x16x32_f16(
                    false, a, false, b, (short)0, acc, false, false);
        }
#pragma unroll
        for (int r = 0; r < 8; ++r) {
            int row = rowbase + r + hh * 8;
            if (row < n) H[(size_t)row * 128 + col0 + ln] = acc[r];
        }
        __syncthreads();
    }
}

// ---------------------------------------------------------------- attention
// one wave per node: as[i] = h[i].a_src ; ad[i] = h[i].a_dst
__global__ void attdot_kernel(const float* __restrict__ h, const float* __restrict__ a_s,
                              const float* __restrict__ a_d, float* __restrict__ os,
                              float* __restrict__ od, int n) {
    int wv = (blockIdx.x * blockDim.x + threadIdx.x) >> 5;
    int lane = threadIdx.x & 31;
    if (wv >= n) return;
    float4 hv = *(const float4*)(h + (size_t)wv * 128 + lane * 4);
    float4 sv = *(const float4*)(a_s + lane * 4);
    float4 dv = *(const float4*)(a_d + lane * 4);
    float s = hv.x * sv.x + hv.y * sv.y + hv.z * sv.z + hv.w * sv.w;
    float d = hv.x * dv.x + hv.y * dv.y + hv.z * dv.z + hv.w * dv.w;
    for (int m = 16; m > 0; m >>= 1) { s += __shfl_xor(s, m, 32); d += __shfl_xor(d, m, 32); }
    if (lane == 0) { os[wv] = s; od[wv] = d; }
}

__device__ __forceinline__ void edge_sd(int e, int E, const int* src, const int* dst,
                                        const int* mask, int& s, int& d, int& m) {
    if (e < E) { s = src[e]; d = dst[e]; m = mask[e]; }
    else       { s = d = e - E; m = 1; }           // self loop
}

// pass 1: per-dst max of leaky_relu logits (encoded-uint atomicMax)
__global__ void edge_max_kernel(const int* __restrict__ src, const int* __restrict__ dst,
                                const int* __restrict__ mask, const float* __restrict__ as_,
                                const float* __restrict__ ad_, unsigned* __restrict__ mx,
                                int E, int n) {
    int e = blockIdx.x * 256 + threadIdx.x;
    if (e >= E + n) return;
    int s, d, m; edge_sd(e, E, src, dst, mask, s, d, m);
    if (!m) return;
    float z = as_[s] + ad_[d];
    float l = z > 0.0f ? z : 0.2f * z;
    atomicMax(mx + d, fenc(l));
}

// pass 2: w = exp(logit - max), accumulate per-dst denom
__global__ void edge_w_kernel(const int* __restrict__ src, const int* __restrict__ dst,
                              const int* __restrict__ mask, const float* __restrict__ as_,
                              const float* __restrict__ ad_, const unsigned* __restrict__ mx,
                              float* __restrict__ denom, float* __restrict__ wexp,
                              int E, int n) {
    int e = blockIdx.x * 256 + threadIdx.x;
    if (e >= E + n) return;
    int s, d, m; edge_sd(e, E, src, dst, mask, s, d, m);
    float w = 0.0f;
    if (m) {
        float z = as_[s] + ad_[d];
        float l = z > 0.0f ? z : 0.2f * z;
        w = expf(l - fdec(mx[d]));
    }
    wexp[e] = w;
    if (w != 0.0f) atomicAdd(denom + d, w);
}

// pass 3: out[d] += alpha * h[s]   (one wave per edge, 32 lanes x float4)
__global__ void edge_scatter_kernel(const int* __restrict__ src, const int* __restrict__ dst,
                                    const float* __restrict__ wexp, const float* __restrict__ denom,
                                    const float* __restrict__ h, float* __restrict__ out,
                                    int E, int n) {
    int e = (blockIdx.x * blockDim.x + threadIdx.x) >> 5;
    int lane = threadIdx.x & 31;
    if (e >= E + n) return;
    int s, d;
    if (e < E) { s = src[e]; d = dst[e]; } else { s = d = e - E; }
    float w = wexp[e];
    if (w == 0.0f) return;
    float alpha = w / denom[d];
    float4 hv = *(const float4*)(h + (size_t)s * 128 + lane * 4);
    float* o = out + (size_t)d * 128 + lane * 4;
    atomicAdd(o + 0, alpha * hv.x);
    atomicAdd(o + 1, alpha * hv.y);
    atomicAdd(o + 2, alpha * hv.z);
    atomicAdd(o + 3, alpha * hv.w);
}

__global__ void bias_gelu_kernel(float* __restrict__ out, const float* __restrict__ b, int n) {
    long long i = (long long)blockIdx.x * 256 + threadIdx.x;
    if (i >= (long long)n * 128) return;
    int c = (int)(i & 127);
    out[i] = gelu_f(out[i] + b[c]);
}

// ---------------------------------------------------------------- pooling
__global__ void norm_kernel(const float* __restrict__ w, float* __restrict__ out) {
    int lane = threadIdx.x & 31;
    float4 v = *(const float4*)(w + lane * 4);
    float s = v.x * v.x + v.y * v.y + v.z * v.z + v.w * v.w;
    for (int m = 16; m > 0; m >>= 1) s += __shfl_xor(s, m, 32);
    if (lane == 0) out[0] = sqrtf(s);
}

__global__ void score_kernel(const float* __restrict__ x, const float* __restrict__ w,
                             const float* __restrict__ nrm, float* __restrict__ score,
                             unsigned* __restrict__ key, int n) {
    int wv = (blockIdx.x * blockDim.x + threadIdx.x) >> 5;
    int lane = threadIdx.x & 31;
    if (wv >= n) return;
    float4 xv = *(const float4*)(x + (size_t)wv * 128 + lane * 4);
    float4 wvv = *(const float4*)(w + lane * 4);
    float s = xv.x * wvv.x + xv.y * wvv.y + xv.z * wvv.z + xv.w * wvv.w;
    for (int m = 16; m > 0; m >>= 1) s += __shfl_xor(s, m, 32);
    if (lane == 0) {
        float t = tanhf(s / nrm[0]);
        score[wv] = t;
        key[wv] = fenc(t);
    }
}

// radix-select: ctl[0]=prefix/T, ctl[1]=remaining, ctl[2]=prefix mask
__global__ void radix_init_kernel(unsigned* ctl, unsigned k) {
    if (threadIdx.x == 0 && blockIdx.x == 0) { ctl[0] = 0u; ctl[1] = k; ctl[2] = 0u; }
}
__global__ void hist_kernel(const unsigned* __restrict__ key, const unsigned* __restrict__ ctl,
                            unsigned* __restrict__ hist, int n, int shift) {
    int i = blockIdx.x * 256 + threadIdx.x;
    if (i >= n) return;
    unsigned kv = key[i];
    if ((kv & ctl[2]) == ctl[0]) atomicAdd(&hist[(kv >> shift) & 255u], 1u);
}
__global__ void pick_kernel(unsigned* __restrict__ ctl, const unsigned* __restrict__ hist,
                            int shift) {
    if (threadIdx.x != 0 || blockIdx.x != 0) return;
    unsigned rem = ctl[1];
    for (int b = 255; b >= 0; --b) {
        unsigned c = hist[b];
        if (rem <= c) { ctl[0] |= ((unsigned)b) << shift; ctl[2] |= 0xFFu << shift; break; }
        rem -= c;
    }
    ctl[1] = rem;
}
__global__ void eq_kernel(const unsigned* __restrict__ key, const unsigned* __restrict__ ctl,
                          unsigned* __restrict__ flag, int n) {
    int i = blockIdx.x * 256 + threadIdx.x;
    if (i < n) flag[i] = (key[i] == ctl[0]) ? 1u : 0u;
}
__global__ void keep_kernel(const unsigned* __restrict__ key, const unsigned* __restrict__ ctl,
                            const unsigned* __restrict__ eqrank, unsigned* __restrict__ keep,
                            int n) {
    int i = blockIdx.x * 256 + threadIdx.x;
    if (i >= n) return;
    unsigned T = ctl[0], rem = ctl[1], kv = key[i];
    keep[i] = (kv > T || (kv == T && eqrank[i] < rem)) ? 1u : 0u;
}

// exclusive scan: 512-wide block scan + serial block-offset scan + add
__global__ void scan_local_kernel(const unsigned* __restrict__ flags, unsigned* __restrict__ ranks,
                                  unsigned* __restrict__ sums, int n) {
    __shared__ unsigned sh[512];
    int i = blockIdx.x * 512 + threadIdx.x;
    unsigned f = (i < n) ? flags[i] : 0u;
    sh[threadIdx.x] = f;
    for (int d = 1; d < 512; d <<= 1) {
        __syncthreads();
        unsigned t = (threadIdx.x >= d) ? sh[threadIdx.x - d] : 0u;
        __syncthreads();
        sh[threadIdx.x] += t;
    }
    __syncthreads();
    if (i < n) ranks[i] = sh[threadIdx.x] - f;
    if (threadIdx.x == 511) sums[blockIdx.x] = sh[511];
}
__global__ void scan_sums_kernel(const unsigned* __restrict__ sums, unsigned* __restrict__ offs,
                                 int nb) {
    if (threadIdx.x != 0 || blockIdx.x != 0) return;
    unsigned run = 0;
    for (int b = 0; b < nb; ++b) { offs[b] = run; run += sums[b]; }
}
__global__ void scan_add_kernel(unsigned* __restrict__ ranks, const unsigned* __restrict__ offs,
                                int n) {
    int i = blockIdx.x * 512 + threadIdx.x;
    if (i < n) ranks[i] += offs[blockIdx.x];
}

// compaction: one wave per kept node (x_new = x * score), remap[i] = new index
__global__ void compact_kernel(const float* __restrict__ xin, const unsigned* __restrict__ keep,
                               const unsigned* __restrict__ pos, const float* __restrict__ score,
                               float* __restrict__ xout, unsigned* __restrict__ remap, int n) {
    int wv = (blockIdx.x * blockDim.x + threadIdx.x) >> 5;
    int lane = threadIdx.x & 31;
    if (wv >= n || !keep[wv]) return;
    unsigned j = pos[wv];
    float sc = score[wv];
    float4 v = *(const float4*)(xin + (size_t)wv * 128 + lane * 4);
    v.x *= sc; v.y *= sc; v.z *= sc; v.w *= sc;
    *(float4*)(xout + (size_t)j * 128 + lane * 4) = v;
    if (lane == 0) remap[wv] = j;
}

__global__ void edge_remap_kernel(const int* __restrict__ src, const int* __restrict__ dst,
                                  const int* __restrict__ mask, const unsigned* __restrict__ keep,
                                  const unsigned* __restrict__ remap, int* __restrict__ nsrc,
                                  int* __restrict__ ndst, int* __restrict__ nmask, int E) {
    int e = blockIdx.x * 256 + threadIdx.x;
    if (e >= E) return;
    int s = src[e], d = dst[e];
    int m = mask[e] && keep[s] && keep[d];
    nsrc[e] = (int)remap[s];
    ndst[e] = (int)remap[d];
    nmask[e] = m ? 1 : 0;
}

// ---------------------------------------------------------------- output head
__global__ void proj_kernel(const float* __restrict__ x, const float* __restrict__ W,
                            const float* __restrict__ b, float* __restrict__ out, int n) {
    int i = blockIdx.x * 256 + threadIdx.x;
    if (i >= n) return;
    float acc[9];
#pragma unroll
    for (int c = 0; c < 9; ++c) acc[c] = b[c];
    const float* xr = x + (size_t)i * 128;
    for (int k = 0; k < 128; ++k) {
        float xv = xr[k];
#pragma unroll
        for (int c = 0; c < 9; ++c) acc[c] += xv * W[k * 9 + c];
    }
#pragma unroll
    for (int c = 0; c < 9; ++c) out[(size_t)i * 9 + c] = acc[c];
}

// ---------------------------------------------------------------- host driver
extern "C" void kernel_launch(void* const* d_in, const int* in_sizes, int n_in,
                              void* d_out, int out_size, void* d_ws, size_t ws_size,
                              hipStream_t stream) {
    (void)in_sizes; (void)n_in; (void)out_size; (void)ws_size;
    // inputs in setup_inputs() insertion order, params depth-first
    const int*   node_types = (const int*)d_in[0];
    const float* positions  = (const float*)d_in[1];
    const int*   edge_src   = (const int*)d_in[2];
    const int*   edge_dst   = (const int*)d_in[3];
    const float* node_emb   = (const float*)d_in[4];
    const float* ln_g       = (const float*)d_in[5];
    const float* ln_b       = (const float*)d_in[6];
    const float* proj_W     = (const float*)d_in[7];
    const float* proj_b     = (const float*)d_in[8];
    struct Gat { const float *W, *as, *ad, *b; };
    Gat enc[3], bot[2];
    for (int i = 0; i < 3; ++i)
        enc[i] = { (const float*)d_in[9 + 4 * i], (const float*)d_in[10 + 4 * i],
                   (const float*)d_in[11 + 4 * i], (const float*)d_in[12 + 4 * i] };
    const float* poolw[3] = { (const float*)d_in[21], (const float*)d_in[22],
                              (const float*)d_in[23] };
    for (int i = 0; i < 2; ++i)
        bot[i] = { (const float*)d_in[24 + 4 * i], (const float*)d_in[25 + 4 * i],
                   (const float*)d_in[26 + 4 * i], (const float*)d_in[27 + 4 * i] };
    const float* out_W = (const float*)d_in[32];
    const float* out_b = (const float*)d_in[33];

    // workspace carve
    char* base = (char*)d_ws; size_t off = 0;
    auto carve = [&](size_t bytes) -> void* {
        void* p = base + off; off += (bytes + 255) & ~(size_t)255; return p;
    };
    float*    B0     = (float*)carve((size_t)NN * 160 * 4);
    float*    B1     = (float*)carve((size_t)NN * 128 * 4);
    float*    B2     = (float*)carve((size_t)NN * 128 * 4);
    float*    as_    = (float*)carve((size_t)NN * 4);
    float*    ad_    = (float*)carve((size_t)NN * 4);
    unsigned* mxenc  = (unsigned*)carve((size_t)NN * 4);
    float*    denom  = (float*)carve((size_t)NN * 4);
    float*    wexp   = (float*)carve((size_t)(NE + NN) * 4);
    float*    score  = (float*)carve((size_t)NN * 4);
    unsigned* key    = (unsigned*)carve((size_t)NN * 4);
    unsigned* flagb  = (unsigned*)carve((size_t)NN * 4);
    unsigned* eqrank = (unsigned*)carve((size_t)NN * 4);
    unsigned* keep   = (unsigned*)carve((size_t)NN * 4);
    unsigned* pos    = (unsigned*)carve((size_t)NN * 4);
    unsigned* remap  = (unsigned*)carve((size_t)NN * 4);
    int* srcA  = (int*)carve((size_t)NE * 4);
    int* dstA  = (int*)carve((size_t)NE * 4);
    int* maskA = (int*)carve((size_t)NE * 4);
    int* srcB  = (int*)carve((size_t)NE * 4);
    int* dstB  = (int*)carve((size_t)NE * 4);
    int* maskB = (int*)carve((size_t)NE * 4);
    unsigned* hist  = (unsigned*)carve(256 * 4);
    unsigned* ctl   = (unsigned*)carve(64);
    unsigned* bsums = (unsigned*)carve(1024 * 4);
    unsigned* boffs = (unsigned*)carve(1024 * 4);
    float*    wnorm = (float*)carve(256);

    auto fill = [&](void* p, unsigned v, long long cnt) {
        fill_u32_kernel<<<(unsigned)((cnt + 255) / 256), 256, 0, stream>>>((unsigned*)p, v, cnt);
    };
    auto scan = [&](const unsigned* flags, unsigned* ranks, int n) {
        int nb = (n + 511) / 512;
        scan_local_kernel<<<nb, 512, 0, stream>>>(flags, ranks, bsums, n);
        scan_sums_kernel<<<1, 32, 0, stream>>>(bsums, boffs, nb);
        scan_add_kernel<<<nb, 512, 0, stream>>>(ranks, boffs, n);
    };
    auto run_gat = [&](const float* x, int ldx, int K, const Gat& g,
                       const int* src, const int* dst, const int* mask,
                       int n, float* h, float* out) {
        fill(mxenc, 0u, n);
        fill(denom, 0u, n);
        fill(out, 0u, (long long)n * 128);
        gemm_wmma_kernel<<<(n + 255) / 256, 256, 0, stream>>>(x, ldx, K, g.W, h, n);
        attdot_kernel<<<(n + 7) / 8, 256, 0, stream>>>(h, g.as, g.ad, as_, ad_, n);
        int total = NE + n;
        edge_max_kernel<<<(total + 255) / 256, 256, 0, stream>>>(src, dst, mask, as_, ad_, mxenc, NE, n);
        edge_w_kernel<<<(total + 255) / 256, 256, 0, stream>>>(src, dst, mask, as_, ad_, mxenc, denom, wexp, NE, n);
        edge_scatter_kernel<<<(total + 7) / 8, 256, 0, stream>>>(src, dst, wexp, denom, h, out, NE, n);
        bias_gelu_kernel<<<(unsigned)(((long long)n * 128 + 255) / 256), 256, 0, stream>>>(out, g.b, n);
    };
    auto run_pool = [&](const float* xin, const float* w, int n, int k, float* xout,
                        const int* src, const int* dst, const int* mask,
                        int* nsrc, int* ndst, int* nmask) {
        norm_kernel<<<1, 32, 0, stream>>>(w, wnorm);
        score_kernel<<<(n + 7) / 8, 256, 0, stream>>>(xin, w, wnorm, score, key, n);
        radix_init_kernel<<<1, 32, 0, stream>>>(ctl, (unsigned)k);
        for (int shift = 24; shift >= 0; shift -= 8) {
            fill(hist, 0u, 256);
            hist_kernel<<<(n + 255) / 256, 256, 0, stream>>>(key, ctl, hist, n, shift);
            pick_kernel<<<1, 32, 0, stream>>>(ctl, hist, shift);
        }
        eq_kernel<<<(n + 255) / 256, 256, 0, stream>>>(key, ctl, flagb, n);
        scan(flagb, eqrank, n);
        keep_kernel<<<(n + 255) / 256, 256, 0, stream>>>(key, ctl, eqrank, keep, n);
        scan(keep, pos, n);
        fill(remap, 0u, n);
        compact_kernel<<<(n + 7) / 8, 256, 0, stream>>>(xin, keep, pos, score, xout, remap, n);
        edge_remap_kernel<<<(NE + 255) / 256, 256, 0, stream>>>(src, dst, mask, keep, remap,
                                                                nsrc, ndst, nmask, NE);
    };

    const int n0 = 300000, n1 = 150000, n2 = 75000, n3 = 37500;

    embed_kernel<<<(n0 + 255) / 256, 256, 0, stream>>>(node_types, positions, node_emb,
                                                       ln_g, ln_b, proj_W, proj_b, B0, n0);
    fill(maskA, 1u, NE);

    // encoder level 0 (EFF_IN=160)
    run_gat(B0, 160, 160, enc[0], edge_src, edge_dst, maskA, n0, B1, B2);
    run_pool(B2, poolw[0], n0, n1, B0, edge_src, edge_dst, maskA, srcB, dstB, maskB);
    // encoder level 1
    run_gat(B0, 128, 128, enc[1], srcB, dstB, maskB, n1, B1, B2);
    run_pool(B2, poolw[1], n1, n2, B0, srcB, dstB, maskB, srcA, dstA, maskA);
    // encoder level 2
    run_gat(B0, 128, 128, enc[2], srcA, dstA, maskA, n2, B1, B2);
    run_pool(B2, poolw[2], n2, n3, B0, srcA, dstA, maskA, srcB, dstB, maskB);
    // bottleneck
    run_gat(B0, 128, 128, bot[0], srcB, dstB, maskB, n3, B1, B2);
    run_gat(B2, 128, 128, bot[1], srcB, dstB, maskB, n3, B1, B0);
    // output head
    proj_kernel<<<(n3 + 255) / 256, 256, 0, stream>>>(B0, out_W, out_b, (float*)d_out, n3);
}
